// NearbyAttention_52372831207513
// MI455X (gfx1250) — matrix operand order repaired and checked by reading, not compiled, not run on hardware
//
#include <hip/hip_runtime.h>

// ---------------------------------------------------------------------------
// NearbyAttention on MI455X (gfx1250, wave32, WMMA 16x16x32 bf16 + TDM)
//   1) Q/K/V projections: double-buffered bf16 WMMA GEMM -> [b,h,tok,d] bf16
//   2) masked flash attention, one wave per (b, h, 16-query tile); V tiles
//      fetched via tensor_load_to_lds (TDM) with OOB zero-fill
//   3) output projection: bf16 WMMA GEMM + bias -> f32
// ---------------------------------------------------------------------------

typedef __attribute__((ext_vector_type(16))) __bf16 v16bf;
typedef __attribute__((ext_vector_type(8)))  float  v8f;
typedef __attribute__((ext_vector_type(4)))  unsigned int u32x4;
typedef __attribute__((ext_vector_type(4)))  int   i32x4;
typedef __attribute__((ext_vector_type(8)))  int   i32x8;

union ABFrag { int i[8]; v16bf v; };   // 8 VGPRs: 16 bf16 per lane
union AccFrag { float f[8]; v8f v; };  // 8 VGPRs f32 C/D

#define NTOK   1025
#define BATCH  16
#define HEADS  16
#define DHEAD  64
#define MTOT   (BATCH * NTOK)          // 16400
#define SCALE  0.125f                  // 64^-0.5
#define NEGINF (-3.0e38f)

#if __has_builtin(__builtin_amdgcn_tensor_load_to_lds)
#define USE_TDM 1
#else
#define USE_TDM 0
#endif

__device__ __forceinline__ unsigned short f2bf(float x) {
  unsigned u = __builtin_bit_cast(unsigned, x);
  u += 0x7FFFu + ((u >> 16) & 1u);     // round-to-nearest-even
  return (unsigned short)(u >> 16);
}

__device__ __forceinline__ void wait_ds0() {
  asm volatile("s_wait_dscnt 0" ::: "memory");
}

#if USE_TDM
// 1-D TDM copy: validElems bf16 from gsrc -> LDS (tile 2048 elems, zero-fill
// past tensor_dim0). D# per CDNA5 ISA ch.8: group0 = {flags, lds_addr,
// global_addr, type=2}; group1 = {data_size=2B, tensor_dim0, tile_dim0}.
__device__ __forceinline__ void tdm_load_1d(const unsigned short* gsrc,
                                            unsigned short* ldst,
                                            unsigned validElems) {
  unsigned long long ga = (unsigned long long)gsrc;
  unsigned lds = (unsigned)(unsigned long long)ldst;
  u32x4 g0;
  g0.x = 1u;                                        // count=1, load, normal
  g0.y = lds;                                       // lds_addr (bytes)
  g0.z = (unsigned)(ga & 0xFFFFFFFFull);            // global_addr[31:0]
  g0.w = (unsigned)((ga >> 32) & 0x01FFFFFFull)     // global_addr[56:32]
         | 0x80000000u;                             // type=2 ("image")
  i32x8 g1;
  g1[0] = (int)(1u << 16);                          // data_size=1 -> 2 bytes
  g1[1] = (int)((validElems & 0xFFFFu) << 16);      // tensor_dim0[15:0]
  g1[2] = (int)((validElems >> 16) & 0xFFFFu);      // tensor_dim0[31:16]
  g1[3] = (int)(2048u << 16);                       // tile_dim0 = 2048 elems
  g1[4] = 0; g1[5] = 0; g1[6] = 0; g1[7] = 0;       // 1-D: dims/strides unused
  i32x4 z4 = {0, 0, 0, 0};
#if defined(__clang_major__) && __clang_major__ >= 23
  i32x8 z8 = {0, 0, 0, 0, 0, 0, 0, 0};
  __builtin_amdgcn_tensor_load_to_lds(g0, g1, z4, z4, z8, 0);
#else
  __builtin_amdgcn_tensor_load_to_lds(g0, g1, z4, z4, 0);
#endif
}
#endif

// ---------------------------------------------------------------------------
// GEMM: D[16400 x 1024] = A[16400 x 1024] * W[1024 x 1024]
// 128x128 block tile, 8 waves (2 M x 4 N), double-buffered LDS: next K-panel
// global loads issue before the WMMA block, LDS stores land after it.
// ---------------------------------------------------------------------------
template <bool A_BF16, bool OUTPROJ>
__global__ __launch_bounds__(256) void gemm_wmma(
    const void* __restrict__ Av, const float* __restrict__ W,
    void* __restrict__ Dst, const float* __restrict__ bias) {
  __shared__ __align__(16) unsigned short ldsA[2][128 * 34]; // [m][k] pad 34
  __shared__ __align__(16) unsigned short ldsB[2][128 * 34]; // [n][k] pad 34

  const int tid  = threadIdx.x;
  const int lane = tid & 31;
  const int wave = tid >> 5;
  const int wm   = (wave >> 2) * 64;
  const int wn   = (wave & 3) * 32;
  const int rowBase = blockIdx.x * 128;
  const int colBase = blockIdx.y * 128;
  const int khalf = lane >> 4;
  const int lm    = lane & 15;
  // staging maps
  const int skk = tid & 31, sr0 = tid >> 5;           // A: rows sr0+8i, col skk
  const int sn  = tid & 127, skb = (tid >> 7) * 16;   // B: col sn, ks skb+i

  const float*          Af = (const float*)Av;
  const unsigned short* Ab = (const unsigned short*)Av;

  AccFrag acc[4][2];
#pragma unroll
  for (int mi = 0; mi < 4; mi++)
#pragma unroll
    for (int ni = 0; ni < 2; ni++)
#pragma unroll
      for (int g = 0; g < 8; g++) acc[mi][ni].f[g] = 0.0f;

  // ---- prologue: stage K-panel 0 into buffer 0 ----
  {
#pragma unroll
    for (int i = 0; i < 16; i++) {
      int ga = rowBase + sr0 + 8 * i;
      if (ga >= MTOT) ga = MTOT - 1;
      unsigned short h = A_BF16 ? Ab[(size_t)ga * 1024 + skk]
                                : f2bf(Af[(size_t)ga * 1024 + skk]);
      ldsA[0][(sr0 + 8 * i) * 34 + skk] = h;
    }
#pragma unroll
    for (int i = 0; i < 16; i++)
      ldsB[0][sn * 34 + skb + i] =
          f2bf(W[(size_t)(skb + i) * 1024 + colBase + sn]);
  }
  __syncthreads();

  int cur = 0;
  for (int k0 = 0; k0 < 1024; k0 += 32, cur ^= 1) {
    const bool hasNext = (k0 + 32) < 1024;
    const int k0n = k0 + 32;

    // ---- issue next-panel global loads into registers ----
    float fA[16]; unsigned short hA[16]; float fB[16];
    if (hasNext) {
#pragma unroll
      for (int i = 0; i < 16; i++) {
        int ga = rowBase + sr0 + 8 * i;
        if (ga >= MTOT) ga = MTOT - 1;
        if (A_BF16) hA[i] = Ab[(size_t)ga * 1024 + k0n + skk];
        else        fA[i] = Af[(size_t)ga * 1024 + k0n + skk];
      }
#pragma unroll
      for (int i = 0; i < 16; i++)
        fB[i] = W[(size_t)(k0n + skb + i) * 1024 + colBase + sn];
      if (k0n + 32 < 1024) {  // speculative prefetch of the panel after next
        int gp = rowBase + sr0;
        if (gp >= MTOT) gp = MTOT - 1;
        size_t off = (size_t)gp * 1024 + k0n + 32 + skk;
        __builtin_prefetch(A_BF16 ? (const void*)(Ab + off)
                                  : (const void*)(Af + off), 0, 1);
      }
    }

    // ---- gather operand fragments from current buffer, WMMA ----
    ABFrag a[4], b[2];
#pragma unroll
    for (int mi = 0; mi < 4; mi++) {
      const unsigned short* base = &ldsA[cur][(wm + mi * 16 + lm) * 34];
#pragma unroll
      for (int p = 0; p < 8; p++) {
        int k = khalf * 16 + 2 * p;
        a[mi].i[p] = *(const int*)(base + k);
      }
    }
#pragma unroll
    for (int ni = 0; ni < 2; ni++) {
      const unsigned short* base = &ldsB[cur][(wn + ni * 16 + lm) * 34];
#pragma unroll
      for (int p = 0; p < 8; p++)
        b[ni].i[p] = *(const int*)(base + khalf * 16 + 2 * p);
    }
#pragma unroll
    for (int mi = 0; mi < 4; mi++)
#pragma unroll
      for (int ni = 0; ni < 2; ni++)
        acc[mi][ni].v = __builtin_amdgcn_wmma_f32_16x16x32_bf16(
            false, a[mi].v, false, b[ni].v, (short)0, acc[mi][ni].v,
            false, false);

    // ---- store next panel into the other buffer ----
    if (hasNext) {
      unsigned short* dA = ldsA[cur ^ 1];
      unsigned short* dB = ldsB[cur ^ 1];
#pragma unroll
      for (int i = 0; i < 16; i++)
        dA[(sr0 + 8 * i) * 34 + skk] = A_BF16 ? hA[i] : f2bf(fA[i]);
#pragma unroll
      for (int i = 0; i < 16; i++)
        dB[sn * 34 + skb + i] = f2bf(fB[i]);
    }
    __syncthreads();
  }

  // ---- epilogue ----
#pragma unroll
  for (int mi = 0; mi < 4; mi++)
#pragma unroll
    for (int ni = 0; ni < 2; ni++)
#pragma unroll
      for (int g = 0; g < 8; g++) {
        int row = rowBase + wm + mi * 16 + g + 8 * khalf;
        int col = colBase + wn + ni * 16 + lm;
        if (row < MTOT) {
          float v = acc[mi][ni].f[g];
          if (OUTPROJ) {
            ((float*)Dst)[(size_t)row * 1024 + col] = v + bias[col];
          } else {
            int bidx = row / NTOK, tok = row % NTOK;
            int h = col >> 6, d = col & 63;
            ((unsigned short*)Dst)[(((size_t)(bidx * HEADS + h)) * NTOK + tok) *
                                       DHEAD + d] = f2bf(v);
          }
        }
      }
}

// ---------------------------------------------------------------------------
// Masked flash attention. One wave per (b, h, 16-query tile); 65 q-tiles.
// Key groups of 32: group 0 (BOS) first, then only groups covering patch
// rows [r-2, r+2]. unmasked(i,j) = j==0 || i==0 || (|dr|<=2 && |dc|<=2).
// ---------------------------------------------------------------------------
__global__ __launch_bounds__(128) void attn_wmma(
    const unsigned short* __restrict__ Qp, const unsigned short* __restrict__ Kp,
    const unsigned short* __restrict__ Vp, unsigned short* __restrict__ AO) {
  __shared__ __align__(16) unsigned short ldsP[4][16 * 32]; // per-wave P tile
  __shared__ __align__(16) unsigned short ldsV[4][32 * 64]; // per-wave V tile

  const int tid  = threadIdx.x;
  const int lane = tid & 31;
  const int wv   = tid >> 5;
  const int khalf = lane >> 4;
  const int lm    = lane & 15;

  const int w  = blockIdx.x * 4 + wv;       // 0..16639
  const int qt = w % 65;
  const int bh = w / 65;
  const int h  = bh & 15, bb = bh >> 4;
  const size_t base = ((size_t)(bb * HEADS + h)) * NTOK * DHEAD;
  const unsigned short* Q  = Qp + base;
  const unsigned short* Kk = Kp + base;
  const unsigned short* Vv = Vp + base;
  const int q0 = qt * 16;

  // Q as A-operand: 2 fragments (d 0..31, 32..63), loaded once
  ABFrag qa[2];
  {
    int tok = q0 + lm; if (tok > 1024) tok = 1024;
    const unsigned short* qrow = Q + (size_t)tok * DHEAD;
#pragma unroll
    for (int half = 0; half < 2; half++)
#pragma unroll
      for (int p = 0; p < 8; p++) {
        int k = half * 32 + khalf * 16 + 2 * p;
        qa[half].i[p] = *(const int*)(qrow + k);
      }
  }

  // per-row (vgpr g) mask precomputation: qi = q0 + g + 8*khalf
  int qri[8], qci[8];
  bool qbos[8], qinv[8];
#pragma unroll
  for (int g = 0; g < 8; g++) {
    int qi = q0 + g + 8 * khalf;
    qbos[g] = (qi == 0);
    qinv[g] = (qi > 1024);
    qri[g] = (qi - 1) >> 5;
    qci[g] = (qi - 1) & 31;
  }

  AccFrag o[4];
  float mrow[8], lrow[8];
#pragma unroll
  for (int s = 0; s < 4; s++)
#pragma unroll
    for (int g = 0; g < 8; g++) o[s].f[g] = 0.0f;
#pragma unroll
  for (int g = 0; g < 8; g++) { mrow[g] = NEGINF; lrow[g] = 0.0f; }

  // key-group range: group 0 + window rows
  int g0, g1;
  if (qt == 0) { g0 = 0; g1 = 32; }             // BOS query -> all keys
  else {
    int piLo = q0 - 1;
    int piHi = q0 + 14; if (piHi > 1023) piHi = 1023;
    int rlo = (piLo >> 5) - 2; if (rlo < 0) rlo = 0;
    int rhi = (piHi >> 5) + 2; if (rhi > 31) rhi = 31;
    int kstart = (rlo == 0) ? 0 : (1 + rlo * 32);
    int kend   = 1 + (rhi + 1) * 32; if (kend > 1025) kend = 1025;
    g0 = kstart >> 5; g1 = (kend - 1) >> 5;
  }
  const int gs = (g0 < 1) ? 1 : g0;
  const int niter = 1 + (g1 - gs + 1);

  AccFrag zz;
#pragma unroll
  for (int g = 0; g < 8; g++) zz.f[g] = 0.0f;

  unsigned short* Pl = ldsP[wv];
  unsigned short* Vl = ldsV[wv];

  for (int it = 0; it < niter; it++) {
    const int kb = ((it == 0) ? 0 : (gs + it - 1)) * 32;

    // ---- V tile (32 tokens x 64) -> LDS: TDM 1-D copy, zero-filled OOB ----
#if USE_TDM
    {
      unsigned valid = (unsigned)(NTOK - kb) * DHEAD;
      if (valid > 2048u) valid = 2048u;
      tdm_load_1d(Vv + (size_t)kb * DHEAD, Vl, valid);
    }
#else
    {
      int tok = kb + lane; if (tok > 1024) tok = 1024;
      const int4* vr = (const int4*)(Vv + (size_t)tok * DHEAD);
      int4* vd = (int4*)(Vl + lane * DHEAD);
#pragma unroll
      for (int i = 0; i < 8; i++) vd[i] = vr[i];
    }
#endif

    // ---- K as B-operand, direct from global: [jt][dhalf] ----
    ABFrag kf[2][2];
#pragma unroll
    for (int jt = 0; jt < 2; jt++) {
      int tok = kb + jt * 16 + lm; if (tok > 1024) tok = 1024;
      const unsigned short* krow = Kk + (size_t)tok * DHEAD;
#pragma unroll
      for (int half = 0; half < 2; half++)
#pragma unroll
        for (int p = 0; p < 8; p++)
          kf[jt][half].i[p] =
              *(const int*)(krow + half * 32 + khalf * 16 + 2 * p);
    }

    // ---- S = Q K^T (16 x 32), f32 accum, chained over d-halves ----
    AccFrag s0, s1;
    s0.v = __builtin_amdgcn_wmma_f32_16x16x32_bf16(false, qa[0].v, false,
                                                   kf[0][0].v, (short)0, zz.v,
                                                   false, false);
    s0.v = __builtin_amdgcn_wmma_f32_16x16x32_bf16(false, qa[1].v, false,
                                                   kf[0][1].v, (short)0, s0.v,
                                                   false, false);
    s1.v = __builtin_amdgcn_wmma_f32_16x16x32_bf16(false, qa[0].v, false,
                                                   kf[1][0].v, (short)0, zz.v,
                                                   false, false);
    s1.v = __builtin_amdgcn_wmma_f32_16x16x32_bf16(false, qa[1].v, false,
                                                   kf[1][1].v, (short)0, s1.v,
                                                   false, false);

    // ---- mask + online softmax; row = g + 8*khalf, cols lm / 16+lm ----
    const int kj0 = kb + lm, kj1 = kj0 + 16;
    const int rj0 = (kj0 - 1) >> 5, cj0 = (kj0 - 1) & 31;
    const int rj1 = (kj1 - 1) >> 5, cj1 = (kj1 - 1) & 31;
    const bool k0ok = (kj0 <= 1024), k1ok = (kj1 <= 1024);
    const bool k0bos = (kj0 == 0);

    float p0[8], p1[8], alpha[8];
#pragma unroll
    for (int g = 0; g < 8; g++) {
      int dr0 = qri[g] - rj0; dr0 = dr0 < 0 ? -dr0 : dr0;
      int dc0 = qci[g] - cj0; dc0 = dc0 < 0 ? -dc0 : dc0;
      int dr1 = qri[g] - rj1; dr1 = dr1 < 0 ? -dr1 : dr1;
      int dc1 = qci[g] - cj1; dc1 = dc1 < 0 ? -dc1 : dc1;
      bool un0 = k0ok && (k0bos ||
                 (!qinv[g] && (qbos[g] || (dr0 <= 2 && dc0 <= 2))));
      bool un1 = k1ok &&
                 (!qinv[g] && (qbos[g] || (dr1 <= 2 && dc1 <= 2)));
      float a = un0 ? s0.f[g] * SCALE : NEGINF;
      float b = un1 ? s1.f[g] * SCALE : NEGINF;
      float r = fmaxf(a, b);
      r = fmaxf(r, __shfl_xor(r, 1));
      r = fmaxf(r, __shfl_xor(r, 2));
      r = fmaxf(r, __shfl_xor(r, 4));
      r = fmaxf(r, __shfl_xor(r, 8));
      float mnew = fmaxf(mrow[g], r);
      float al   = __expf(mrow[g] - mnew);
      float e0   = __expf(a - mnew);
      float e1   = __expf(b - mnew);
      float rs = e0 + e1;
      rs += __shfl_xor(rs, 1);
      rs += __shfl_xor(rs, 2);
      rs += __shfl_xor(rs, 4);
      rs += __shfl_xor(rs, 8);
      mrow[g] = mnew;
      lrow[g] = lrow[g] * al + rs;
      alpha[g] = al;
      p0[g] = e0; p1[g] = e1;
    }
#pragma unroll
    for (int s = 0; s < 4; s++)
#pragma unroll
      for (int g = 0; g < 8; g++) o[s].f[g] *= alpha[g];

    // ---- stage P (16x32 bf16 row-major) in LDS ----
#pragma unroll
    for (int g = 0; g < 8; g++) {
      int row = g + 8 * khalf;
      Pl[row * 32 + lm]      = f2bf(p0[g]);
      Pl[row * 32 + 16 + lm] = f2bf(p1[g]);
    }
#if USE_TDM
    __builtin_amdgcn_s_wait_tensorcnt(0);  // V tile landed in LDS
#endif
    wait_ds0();  // same-wave P stores visible for cross-lane gather

    // ---- O += P * V : P as A-operand (K=32 keys), V as B per d subtile ----
    ABFrag pf;
#pragma unroll
    for (int p = 0; p < 8; p++) {
      int k = khalf * 16 + 2 * p;
      pf.i[p] = *(const int*)(Pl + lm * 32 + k);
    }
#pragma unroll
    for (int sub = 0; sub < 4; sub++) {
      ABFrag vf;
#pragma unroll
      for (int p = 0; p < 8; p++) {
        int k = khalf * 16 + 2 * p;
        unsigned lo = Vl[k * DHEAD + sub * 16 + lm];
        unsigned hi = Vl[(k + 1) * DHEAD + sub * 16 + lm];
        vf.i[p] = (int)(lo | (hi << 16));
      }
      o[sub].v = __builtin_amdgcn_wmma_f32_16x16x32_bf16(
          false, pf.v, false, vf.v, (short)0, o[sub].v, false, false);
    }
    wait_ds0();  // drain LDS reads before next group's TDM/stores overwrite
  }

  // ---- normalize and store bf16 into AO[b][tok][h*64+d] ----
#pragma unroll
  for (int sub = 0; sub < 4; sub++)
#pragma unroll
    for (int g = 0; g < 8; g++) {
      int tok = q0 + g + 8 * khalf;
      if (tok < NTOK) {
        float v = o[sub].f[g] / lrow[g];
        AO[((size_t)bb * NTOK + tok) * 1024 + h * DHEAD + sub * 16 + lm] =
            f2bf(v);
      }
    }
}

// ---------------------------------------------------------------------------
extern "C" void kernel_launch(void* const* d_in, const int* in_sizes, int n_in,
                              void* d_out, int out_size, void* d_ws,
                              size_t ws_size, hipStream_t stream) {
  (void)in_sizes; (void)n_in; (void)out_size; (void)ws_size;
  const float* q  = (const float*)d_in[0];
  const float* k  = (const float*)d_in[1];
  const float* v  = (const float*)d_in[2];
  const float* Wq = (const float*)d_in[3];
  const float* Wk = (const float*)d_in[4];
  const float* Wv = (const float*)d_in[5];
  const float* Wo = (const float*)d_in[6];
  const float* bo = (const float*)d_in[7];
  float* out = (float*)d_out;

  char* ws = (char*)d_ws;
  const size_t szP = (size_t)BATCH * HEADS * NTOK * DHEAD * 2;  // 33.6 MB
  unsigned short* Qp = (unsigned short*)ws;
  unsigned short* Kp = (unsigned short*)(ws + szP);
  unsigned short* Vp = (unsigned short*)(ws + 2 * szP);
  unsigned short* AO = (unsigned short*)(ws + 3 * szP);

  dim3 gGrid(129, 8, 1);  // ceil(16400/128) x (1024/128)
  gemm_wmma<false, false><<<gGrid, 256, 0, stream>>>(q, Wq, Qp, nullptr);
  gemm_wmma<false, false><<<gGrid, 256, 0, stream>>>(k, Wk, Kp, nullptr);
  gemm_wmma<false, false><<<gGrid, 256, 0, stream>>>(v, Wv, Vp, nullptr);
  attn_wmma<<<dim3(4160), 128, 0, stream>>>(Qp, Kp, Vp, AO);  // 16*16*65 waves
  gemm_wmma<true, true><<<gGrid, 256, 0, stream>>>(AO, Wo, out, bo);
}
